// STA_Block_72069551227131
// MI455X (gfx1250) — matrix operand order repaired and checked by reading, not compiled, not run on hardware
//
#include <hip/hip_runtime.h>
#include <hip/hip_bf16.h>

typedef _Float16 h16;
typedef __attribute__((ext_vector_type(16))) _Float16 v16h;
typedef __attribute__((ext_vector_type(8)))  _Float16 v8h;
typedef __attribute__((ext_vector_type(8)))  float    v8f;

#define S_SP   3200      // T*V
#define T_DIM  128
#define V_DIM  25
#define H_N    3

#define WMMA_F16(a, b, c) \
  __builtin_amdgcn_wmma_f32_16x16x32_f16(false, (a), false, (b), (short)0, (c), false, false)

__device__ __forceinline__ v16h make_frag(v8h lo, v8h hi) {
  v16h r;
  #pragma unroll
  for (int j = 0; j < 8; ++j) { r[j] = lo[j]; r[8 + j] = hi[j]; }
  return r;
}

// ---------------- prep kernels ----------------

__global__ void cast_f32_f16_kernel(const float* __restrict__ src, h16* __restrict__ dst, long n) {
  long i = (long)blockIdx.x * blockDim.x + threadIdx.x;
  const long stride = (long)gridDim.x * blockDim.x;
  for (; i < n; i += stride) dst[i] = (h16)src[i];
}

// xb plain f16 copy; xpad[n][(c*128+t)*32 + v] zero-padded K rows for xs GEMM
__global__ void cast_x_kernel(const float* __restrict__ x, h16* __restrict__ xb,
                              h16* __restrict__ xpad) {
  const long n_x = 128L * 64 * S_SP;
  const long stride = (long)gridDim.x * blockDim.x;
  for (long i = (long)blockIdx.x * blockDim.x + threadIdx.x; i < n_x; i += stride)
    xb[i] = (h16)x[i];
  const long n_p = 128L * 8192 * 32;
  for (long j = (long)blockIdx.x * blockDim.x + threadIdx.x; j < n_p; j += stride) {
    const long r = j >> 5; const int col = (int)(j & 31);
    xpad[j] = (col < V_DIM) ? (h16)x[r * V_DIM + col] : (h16)0.f;
  }
}

// w_t[o][i][dt] -> wtr[dt][o][i]
__global__ void repack_wt_kernel(const float* __restrict__ wt, h16* __restrict__ wtr) {
  const long n = 128L * 128 * 7;
  const long stride = (long)gridDim.x * blockDim.x;
  for (long idx = (long)blockIdx.x * blockDim.x + threadIdx.x; idx < n; idx += stride) {
    const int o = (int)(idx / 896), rem = (int)(idx % 896);
    const int i = rem / 7, dt = rem % 7;
    wtr[((long)dt * 128 + o) * 128 + i] = (h16)wt[idx];
  }
}

__global__ void bnprep_kernel(const float* __restrict__ bias, const float* __restrict__ g,
                              const float* __restrict__ be, const float* __restrict__ mean,
                              const float* __restrict__ var, float* __restrict__ scale,
                              float* __restrict__ shift, int n) {
  const int i = blockIdx.x * blockDim.x + threadIdx.x;
  if (i < n) {
    const float sc = g[i] * rsqrtf(var[i] + 1e-5f);
    scale[i] = sc;
    shift[i] = (bias[i] - mean[i]) * sc + be[i];
  }
}

// ---------------- 16x16-per-wave GEMM (qkv only, M=96) ----------------
template<int K, bool LRELU>
__global__ void gemm_bn_kernel(const h16* __restrict__ W,
                               const h16* __restrict__ Bbase, long bstride,
                               const float* __restrict__ scale, const float* __restrict__ shift,
                               const h16* __restrict__ Rbase, long rstride,
                               h16* __restrict__ Obase, long ostride) {
  constexpr int LDT = 40;
  __shared__ h16 Bt[64 * LDT];
  const int n    = blockIdx.z;
  const h16* Bn  = Bbase + (long)n * bstride;
  const int tid  = (int)threadIdx.x;
  const int wave = tid >> 5, l = tid & 31;
  const int m0    = blockIdx.y * 32 + (wave >> 2) * 16;
  const int s0blk = blockIdx.x * 64;
  const int sl    = (wave & 3) * 16 + (l & 15);
  const int kb    = (l & 16) ? 16 : 0;
  const int kh    = (l & 16) ? 16 : 0;
  const int mA    = m0 + (l & 15);
  const int kst   = tid >> 3;
  const int sst   = (tid & 7) * 8;
  v8f acc = {};
  #pragma unroll
  for (int k0 = 0; k0 < K; k0 += 32) {
    const v8h bv = *(const v8h*)(Bn + (long)(k0 + kst) * S_SP + s0blk + sst);
    __syncthreads();
    #pragma unroll
    for (int j = 0; j < 8; ++j) Bt[(sst + j) * LDT + kst] = bv[j];
    __syncthreads();
    if (k0 + 32 < K)
      __builtin_prefetch((const void*)(Bn + (long)(k0 + 32 + kst) * S_SP + s0blk + sst), 0, 1);
    const h16* row = W + (long)mA * K + k0 + kb;
    const v16h a = make_frag(*(const v8h*)row, *(const v8h*)(row + 8));
    const v16h b = make_frag(*(const v8h*)&Bt[sl * LDT + kh],
                             *(const v8h*)&Bt[sl * LDT + kh + 8]);
    acc = WMMA_F16(a, b, acc);
  }
  const int s  = s0blk + sl;
  const int mb = m0 + ((l & 16) ? 8 : 0);
  #pragma unroll
  for (int r = 0; r < 8; ++r) {
    const int m = mb + r;
    float y = acc[r];
    if (scale) y *= scale[m];
    if (shift) y += shift[m];
    if (Rbase) y += (float)Rbase[(long)n * rstride + (long)m * S_SP + s];
    if (LRELU) y = (y >= 0.f) ? y : 0.1f * y;
    Obase[(long)n * ostride + (long)m * S_SP + s] = (h16)y;
  }
}

// ---------------- 32x32-per-wave GEMM (M=128), block 64M x 128S ----------------
template<int K, bool LRELU>
__global__ void gemm32_bn_kernel(const h16* __restrict__ W,
                                 const h16* __restrict__ Bbase, long bstride,
                                 const float* __restrict__ scale, const float* __restrict__ shift,
                                 const h16* __restrict__ Rbase, long rstride,
                                 h16* __restrict__ Obase, long ostride) {
  constexpr int LDT = 40;
  __shared__ h16 Bt[128 * LDT];                  // 5120 halves, 10 KB
  const int n    = blockIdx.z;
  const h16* Bn  = Bbase + (long)n * bstride;
  const int tid  = (int)threadIdx.x;
  const int wave = tid >> 5, l = tid & 31;
  const int m0    = blockIdx.y * 64 + (wave >> 2) * 32;
  const int s0blk = blockIdx.x * 128;
  const int slb   = (wave & 3) * 32;
  const int kb    = (l & 16) ? 16 : 0;
  const int kh    = (l & 16) ? 16 : 0;
  const int mA0   = m0 + (l & 15);
  const int sl0   = slb + (l & 15);
  const int kst   = tid >> 3;                    // staging row 0..31
  const int sst   = (tid & 7) * 16;              // staging cols (16 per thread)
  v8f acc[2][2] = {};
  #pragma unroll
  for (int k0 = 0; k0 < K; k0 += 32) {
    const h16* gb = Bn + (long)(k0 + kst) * S_SP + s0blk + sst;
    const v8h bv0 = *(const v8h*)(gb);
    const v8h bv1 = *(const v8h*)(gb + 8);
    __syncthreads();
    #pragma unroll
    for (int j = 0; j < 8; ++j) {
      Bt[(sst + j) * LDT + kst]     = bv0[j];
      Bt[(sst + 8 + j) * LDT + kst] = bv1[j];
    }
    __syncthreads();
    if (k0 + 32 < K)
      __builtin_prefetch((const void*)(Bn + (long)(k0 + 32 + kst) * S_SP + s0blk + sst), 0, 1);
    const h16* r0 = W + (long)mA0 * K + k0 + kb;
    const h16* r1 = r0 + 16 * K;
    const v16h a0 = make_frag(*(const v8h*)r0, *(const v8h*)(r0 + 8));
    const v16h a1 = make_frag(*(const v8h*)r1, *(const v8h*)(r1 + 8));
    const v16h b0 = make_frag(*(const v8h*)&Bt[sl0 * LDT + kh],
                              *(const v8h*)&Bt[sl0 * LDT + kh + 8]);
    const v16h b1 = make_frag(*(const v8h*)&Bt[(sl0 + 16) * LDT + kh],
                              *(const v8h*)&Bt[(sl0 + 16) * LDT + kh + 8]);
    acc[0][0] = WMMA_F16(a0, b0, acc[0][0]);
    acc[0][1] = WMMA_F16(a0, b1, acc[0][1]);
    acc[1][0] = WMMA_F16(a1, b0, acc[1][0]);
    acc[1][1] = WMMA_F16(a1, b1, acc[1][1]);
  }
  const int mb = m0 + ((l & 16) ? 8 : 0);
  #pragma unroll
  for (int mi = 0; mi < 2; ++mi) {
    #pragma unroll
    for (int si = 0; si < 2; ++si) {
      const int s = s0blk + slb + si * 16 + (l & 15);
      #pragma unroll
      for (int r = 0; r < 8; ++r) {
        const int m = mb + mi * 16 + r;
        float y = acc[mi][si][r];
        if (scale) y *= scale[m];
        if (shift) y += shift[m];
        if (Rbase) y += (float)Rbase[(long)n * rstride + (long)m * S_SP + s];
        if (LRELU) y = (y >= 0.f) ? y : 0.1f * y;
        Obase[(long)n * ostride + (long)m * S_SP + s] = (h16)y;
      }
    }
  }
}

// ---------------- attention: logits + softmax (LDS-staged q/k), transposed output ----------------
__global__ void att_kernel(const h16* __restrict__ qk, const float* __restrict__ values,
                           const float* __restrict__ att_bias, h16* __restrict__ attT) {
  const int h = blockIdx.x, n = blockIdx.y;
  const h16* qbase = qk + ((long)n * 96 + h * 16) * S_SP;
  const h16* kbase = qk + ((long)n * 96 + 48 + h * 16) * S_SP;
  __shared__ h16  qs[16 * 800];
  __shared__ h16  ks[16 * 800];
  __shared__ float attm[625];
  float acc[3] = {0.f, 0.f, 0.f};
  const int tid = (int)threadIdx.x;
  for (int c = 0; c < 4; ++c) {
    for (int idx = tid; idx < 16 * 800; idx += 256) {
      const int d = idx / 800, s = idx % 800;
      qs[idx] = qbase[(long)d * S_SP + c * 800 + s];
      ks[idx] = kbase[(long)d * S_SP + c * 800 + s];
    }
    __syncthreads();
    #pragma unroll
    for (int pi = 0; pi < 3; ++pi) {
      const int p = tid + pi * 256;
      if (p < 625) {
        const int u = p / 25, v = p % 25;
        float sacc = acc[pi];
        for (int d = 0; d < 16; ++d) {
          const h16* qr = qs + d * 800;
          const h16* kr = ks + d * 800;
          #pragma unroll 8
          for (int t = 0; t < 32; ++t)
            sacc += (float)qr[t * 25 + u] * (float)kr[t * 25 + v];
        }
        acc[pi] = sacc;
      }
    }
    __syncthreads();
  }
  const float sc = values[h] / 2048.f;
  const float ab = att_bias[h];
  #pragma unroll
  for (int pi = 0; pi < 3; ++pi) {
    const int p = tid + pi * 256;
    if (p < 625) attm[p] = acc[pi] * sc + ab;
  }
  __syncthreads();
  if (tid < 25) {
    float mx = -1e30f;
    for (int v = 0; v < 25; ++v) mx = fmaxf(mx, attm[tid * 25 + v]);
    float e[25], sum = 0.f;
    for (int v = 0; v < 25; ++v) { e[v] = __expf(attm[tid * 25 + v] - mx); sum += e[v]; }
    const float inv = 1.f / sum;
    for (int v = 0; v < 25; ++v) attm[tid * 25 + v] = e[v] * inv;
  }
  __syncthreads();
  h16* out = attT + ((long)n * H_N + h) * 1024;  // attT[v][u], zero-padded 32x32
  for (int idx = tid; idx < 1024; idx += 256) {
    const int v = idx >> 5, u = idx & 31;
    out[idx] = (v < 25 && u < 25) ? (h16)attm[u * 25 + v] : (h16)0.f;
  }
}

// ---------------- xs = x @ att : per-(n,h), M=8192 (c,t), K=32 (padded), N=32 ----------------
__global__ void xs_kernel(const h16* __restrict__ xpad, const h16* __restrict__ attT,
                          h16* __restrict__ xs) {
  const int n = blockIdx.z, h = blockIdx.y;
  const int tid  = (int)threadIdx.x;
  const int wave = tid >> 5, l = tid & 31;       // 4 waves: 2(M) x 2(V)
  const int m0 = blockIdx.x * 32 + (wave >> 1) * 16;
  const int s0 = (wave & 1) * 16;
  const int kb = (l & 16) ? 16 : 0;
  const int kh = (l & 16) ? 16 : 0;
  const h16* Arow = xpad + (long)n * (8192L * 32) + (long)(m0 + (l & 15)) * 32 + kb;
  const h16* Bcol = attT + ((long)n * H_N + h) * 1024 + (long)(s0 + (l & 15)) * 32 + kh;
  const v16h a = make_frag(*(const v8h*)Arow, *(const v8h*)(Arow + 8));
  const v16h b = make_frag(*(const v8h*)Bcol, *(const v8h*)(Bcol + 8));
  v8f acc = {};
  acc = WMMA_F16(a, b, acc);
  const int v = s0 + (l & 15);
  if (v < V_DIM) {
    #pragma unroll
    for (int r = 0; r < 8; ++r) {
      const int m = m0 + ((l & 16) ? 8 : 0) + r;   // row = c*128 + t
      const int c = m >> 7, t = m & 127;
      xs[(long)n * (192L * S_SP) + (long)(h * 64 + c) * S_SP + t * 25 + v] = (h16)acc[r];
    }
  }
}

// ---------------- final: temporal conv (7 taps, halo-staged) + rest, 32x32 per wave ----------------
__global__ void final_kernel(const h16* __restrict__ wtr,   // [7][128][128]
                             const h16* __restrict__ wrest,
                             const h16* __restrict__ x2,
                             const float* __restrict__ st, const float* __restrict__ sht,
                             const float* __restrict__ sr, const float* __restrict__ shr,
                             float* __restrict__ out) {
  constexpr int HALO = 75;             // 3 * 25
  constexpr int SW   = 128 + 2 * HALO; // 278 staged columns
  constexpr int LDT  = 40;
  __shared__ h16 Bt[SW * LDT];         // ~22 KB
  const int n    = blockIdx.z;
  const h16* Bn  = x2 + (long)n * (128L * S_SP);
  const int tid  = (int)threadIdx.x;
  const int wave = tid >> 5, l = tid & 31;
  const int m0    = blockIdx.y * 64 + (wave >> 2) * 32;
  const int s0blk = blockIdx.x * 128;
  const int slb   = (wave & 3) * 32;
  const int kb    = (l & 16) ? 16 : 0;
  const int kh    = (l & 16) ? 16 : 0;
  const int mA0   = m0 + (l & 15);
  const int sl0   = slb + (l & 15);
  const int sl1   = sl0 + 16;
  const int t0    = (s0blk + sl0) / V_DIM;
  const int t1    = (s0blk + sl1) / V_DIM;
  v8f acct[2][2] = {};
  v8f accr[2][2] = {};
  #pragma unroll
  for (int k0 = 0; k0 < 128; k0 += 32) {
    __syncthreads();
    for (int idx = tid; idx < 32 * SW; idx += 256) {
      const int kk = idx / SW, scc = idx % SW;
      const int gs = s0blk - HALO + scc;
      h16 val = (h16)0.f;
      if (gs >= 0 && gs < S_SP) val = Bn[(long)(k0 + kk) * S_SP + gs];
      Bt[scc * LDT + kk] = val;
    }
    __syncthreads();
    #pragma unroll
    for (int dt = 0; dt < 7; ++dt) {
      const h16* r0 = wtr + (((long)dt * 128 + mA0) * 128) + k0 + kb;
      const h16* r1 = r0 + 16 * 128;
      const v16h a0 = make_frag(*(const v8h*)r0, *(const v8h*)(r0 + 8));
      const v16h a1 = make_frag(*(const v8h*)r1, *(const v8h*)(r1 + 8));
      v16h b0 = {}, b1 = {};
      const int sh25 = HALO + V_DIM * (dt - 3);
      const int ts0 = t0 + dt - 3, ts1 = t1 + dt - 3;
      if (ts0 >= 0 && ts0 < T_DIM) {
        const int li = sl0 + sh25;
        b0 = make_frag(*(const v8h*)&Bt[li * LDT + kh], *(const v8h*)&Bt[li * LDT + kh + 8]);
      }
      if (ts1 >= 0 && ts1 < T_DIM) {
        const int li = sl1 + sh25;
        b1 = make_frag(*(const v8h*)&Bt[li * LDT + kh], *(const v8h*)&Bt[li * LDT + kh + 8]);
      }
      acct[0][0] = WMMA_F16(a0, b0, acct[0][0]);
      acct[0][1] = WMMA_F16(a0, b1, acct[0][1]);
      acct[1][0] = WMMA_F16(a1, b0, acct[1][0]);
      acct[1][1] = WMMA_F16(a1, b1, acct[1][1]);
    }
    {
      const h16* r0 = wrest + (long)mA0 * 128 + k0 + kb;
      const h16* r1 = r0 + 16 * 128;
      const v16h a0 = make_frag(*(const v8h*)r0, *(const v8h*)(r0 + 8));
      const v16h a1 = make_frag(*(const v8h*)r1, *(const v8h*)(r1 + 8));
      const int li0 = sl0 + HALO, li1 = sl1 + HALO;
      const v16h b0 = make_frag(*(const v8h*)&Bt[li0 * LDT + kh], *(const v8h*)&Bt[li0 * LDT + kh + 8]);
      const v16h b1 = make_frag(*(const v8h*)&Bt[li1 * LDT + kh], *(const v8h*)&Bt[li1 * LDT + kh + 8]);
      accr[0][0] = WMMA_F16(a0, b0, accr[0][0]);
      accr[0][1] = WMMA_F16(a0, b1, accr[0][1]);
      accr[1][0] = WMMA_F16(a1, b0, accr[1][0]);
      accr[1][1] = WMMA_F16(a1, b1, accr[1][1]);
    }
  }
  const int mb = m0 + ((l & 16) ? 8 : 0);
  #pragma unroll
  for (int mi = 0; mi < 2; ++mi) {
    #pragma unroll
    for (int si = 0; si < 2; ++si) {
      const int s = s0blk + slb + si * 16 + (l & 15);
      #pragma unroll
      for (int r = 0; r < 8; ++r) {
        const int m = mb + mi * 16 + r;
        float y = acct[mi][si][r] * st[m] + sht[m] + accr[mi][si][r] * sr[m] + shr[m];
        y = (y >= 0.f) ? y : 0.1f * y;
        out[(long)n * (128L * S_SP) + (long)m * S_SP + s] = y;
      }
    }
  }
}

// ---------------- host launcher ----------------
extern "C" void kernel_launch(void* const* d_in, const int* in_sizes, int n_in,
                              void* d_out, int out_size, void* d_ws, size_t ws_size,
                              hipStream_t stream) {
  (void)in_sizes; (void)n_in; (void)out_size; (void)ws_size;
  const float* x       = (const float*)d_in[0];
  const float* w_qkv   = (const float*)d_in[1];
  const float* b_qkv   = (const float*)d_in[2];
  const float* values  = (const float*)d_in[3];
  const float* att_b   = (const float*)d_in[4];
  const float* w_out   = (const float*)d_in[5];
  const float* w_ff    = (const float*)d_in[11];
  const float* w_t     = (const float*)d_in[17];
  const float* w_ress  = (const float*)d_in[23];
  const float* w_rest  = (const float*)d_in[29];
  float* out = (float*)d_out;

  char* ws = (char*)d_ws;
  size_t off = 0;
  auto alloc = [&](size_t bytes) { size_t cur = off; off = (off + bytes + 255) & ~(size_t)255; return cur; };

  const long n_x = 128L * 64 * S_SP;
  h16* xb     = (h16*)(ws + alloc(n_x * 2));
  h16* xpad   = (h16*)(ws + alloc(128L * 8192 * 32 * 2));
  h16* wqkv_h = (h16*)(ws + alloc(96 * 64 * 2));
  h16* wout_h = (h16*)(ws + alloc(128 * 192 * 2));
  h16* wff_h  = (h16*)(ws + alloc(128 * 128 * 2));
  h16* wtr_h  = (h16*)(ws + alloc(128 * 128 * 7 * 2));
  h16* wress_h= (h16*)(ws + alloc(128 * 64 * 2));
  h16* wrest_h= (h16*)(ws + alloc(128 * 128 * 2));
  h16* qk     = (h16*)(ws + alloc(128L * 96 * S_SP * 2));
  h16* attT   = (h16*)(ws + alloc(128L * H_N * 1024 * 2));
  h16* xs     = (h16*)(ws + alloc(128L * 192 * S_SP * 2));
  h16* xr     = (h16*)(ws + alloc(128L * 128 * S_SP * 2));
  h16* x1     = (h16*)(ws + alloc(128L * 128 * S_SP * 2));
  h16* x2     = (h16*)(ws + alloc(128L * 128 * S_SP * 2));
  float* s_out  = (float*)(ws + alloc(128 * 4)); float* sh_out  = (float*)(ws + alloc(128 * 4));
  float* s_ff   = (float*)(ws + alloc(128 * 4)); float* sh_ff   = (float*)(ws + alloc(128 * 4));
  float* s_t    = (float*)(ws + alloc(128 * 4)); float* sh_t    = (float*)(ws + alloc(128 * 4));
  float* s_ress = (float*)(ws + alloc(128 * 4)); float* sh_ress = (float*)(ws + alloc(128 * 4));
  float* s_rest = (float*)(ws + alloc(128 * 4)); float* sh_rest = (float*)(ws + alloc(128 * 4));

  cast_x_kernel<<<4096, 256, 0, stream>>>(x, xb, xpad);
  cast_f32_f16_kernel<<<24, 256, 0, stream>>>(w_qkv, wqkv_h, 96 * 64);
  cast_f32_f16_kernel<<<96, 256, 0, stream>>>(w_out, wout_h, 128 * 192);
  cast_f32_f16_kernel<<<64, 256, 0, stream>>>(w_ff, wff_h, 128 * 128);
  repack_wt_kernel<<<448, 256, 0, stream>>>(w_t, wtr_h);
  cast_f32_f16_kernel<<<32, 256, 0, stream>>>(w_ress, wress_h, 128 * 64);
  cast_f32_f16_kernel<<<64, 256, 0, stream>>>(w_rest, wrest_h, 128 * 128);

  bnprep_kernel<<<1, 128, 0, stream>>>((const float*)d_in[6],  (const float*)d_in[7],
                                       (const float*)d_in[8],  (const float*)d_in[9],
                                       (const float*)d_in[10], s_out, sh_out, 128);
  bnprep_kernel<<<1, 128, 0, stream>>>((const float*)d_in[12], (const float*)d_in[13],
                                       (const float*)d_in[14], (const float*)d_in[15],
                                       (const float*)d_in[16], s_ff, sh_ff, 128);
  bnprep_kernel<<<1, 128, 0, stream>>>((const float*)d_in[18], (const float*)d_in[19],
                                       (const float*)d_in[20], (const float*)d_in[21],
                                       (const float*)d_in[22], s_t, sh_t, 128);
  bnprep_kernel<<<1, 128, 0, stream>>>((const float*)d_in[24], (const float*)d_in[25],
                                       (const float*)d_in[26], (const float*)d_in[27],
                                       (const float*)d_in[28], s_ress, sh_ress, 128);
  bnprep_kernel<<<1, 128, 0, stream>>>((const float*)d_in[30], (const float*)d_in[31],
                                       (const float*)d_in[32], (const float*)d_in[33],
                                       (const float*)d_in[34], s_rest, sh_rest, 128);

  const long bx = 64L * S_SP, b96 = 96L * S_SP, b128 = 128L * S_SP, b192 = 192L * S_SP;

  // qkv = w_qkv @ x + b_qkv  (M=96, 16x16 wave tile)
  gemm_bn_kernel<64, false><<<dim3(50, 3, 128), 256, 0, stream>>>(
      wqkv_h, xb, bx, nullptr, b_qkv, nullptr, 0, qk, b96);

  att_kernel<<<dim3(H_N, 128), 256, 0, stream>>>(qk, values, att_b, attT);

  xs_kernel<<<dim3(256, H_N, 128), 128, 0, stream>>>(xpad, attT, xs);

  // xr = bn_ress(w_ress @ x)
  gemm32_bn_kernel<64, false><<<dim3(25, 2, 128), 256, 0, stream>>>(
      wress_h, xb, bx, s_ress, sh_ress, nullptr, 0, xr, b128);

  // x1 = lrelu(bn_out(w_out @ xs) + xr)
  gemm32_bn_kernel<192, true><<<dim3(25, 2, 128), 256, 0, stream>>>(
      wout_h, xs, b192, s_out, sh_out, xr, b128, x1, b128);

  // x2 = lrelu(bn_ff(w_ff @ x1) + xr)
  gemm32_bn_kernel<128, true><<<dim3(25, 2, 128), 256, 0, stream>>>(
      wff_h, x1, b128, s_ff, sh_ff, xr, b128, x2, b128);

  // out = lrelu(bn_t(tconv7(x2)) + bn_rest(w_rest @ x2))
  final_kernel<<<dim3(25, 2, 128), 256, 0, stream>>>(
      wtr_h, wrest_h, x2, s_t, sh_t, s_rest, sh_rest, out);
}
